// KleinManifold_15393162789091
// MI455X (gfx1250) — matrix operand order repaired and checked by reading, not compiled, not run on hardware
//
#include <hip/hip_runtime.h>

typedef __attribute__((ext_vector_type(16))) _Float16 v16h;
typedef __attribute__((ext_vector_type(8)))  _Float16 v8h;
typedef __attribute__((ext_vector_type(4)))  _Float16 v4h;
typedef __attribute__((ext_vector_type(8)))  float    v8f;
typedef __attribute__((ext_vector_type(4)))  float    v4f;

#define DIN   1024
#define NROWS 65536
#define ROWS  64          // rows of x per block (4 x 16-row WMMA tiles)
#define RT    4           // row tiles per block

__device__ __forceinline__ float wave_reduce_sum(float v) {
    #pragma unroll
    for (int off = 16; off > 0; off >>= 1)
        v += __shfl_xor(v, off, 32);
    return v;
}

// ---------------------------------------------------------------------------
// K0: stabilize m rows (add 1e-6 to col 0 of tiny-norm rows), convert to f16.
// One block per row of m.
// ---------------------------------------------------------------------------
__global__ void __launch_bounds__(256)
prep_m(const float* __restrict__ m, _Float16* __restrict__ mh) {
    __shared__ float red[256];
    const int j = blockIdx.x;
    const int t = threadIdx.x;
    const float* row = m + (size_t)j * DIN;

    float s = 0.f;
    for (int k = t; k < DIN; k += 256) { float v = row[k]; s += v * v; }
    red[t] = s;
    __syncthreads();
    #pragma unroll
    for (int o = 128; o > 0; o >>= 1) {
        if (t < o) red[t] += red[t + o];
        __syncthreads();
    }
    const float fix = (sqrtf(red[0]) < 1e-8f) ? 1e-6f : 0.f;

    for (int k = t; k < DIN; k += 256) {
        float v = row[k] + ((k == 0) ? fix : 0.f);
        mh[(size_t)j * DIN + k] = (_Float16)v;
    }
}

// ---------------------------------------------------------------------------
// K1: fully fused: logmap0(x) -> f16 LDS tile, WMMA GEMM mx = x_tan @ m^T
// (clip +-1000), per-row sum-of-squares, then expmap0/proj rescale applied
// in-place (block owns complete rows, so the row norm resolves locally).
// Block = 256 threads (8 waves), owns 64 rows x all 1024 output cols.
// ---------------------------------------------------------------------------
__global__ void __launch_bounds__(256)
mobius_gemm(const float* __restrict__ x, const _Float16* __restrict__ mh,
            float* __restrict__ out) {
    __shared__ __attribute__((aligned(16))) _Float16 xt[ROWS * DIN]; // 128 KB
    __shared__ float ssq[ROWS];

    const int tid  = threadIdx.x;
    const int lane = tid & 31;
    const int wave = tid >> 5;
    const int l15  = lane & 15;
    const int h16  = lane >> 4;          // 0 for lanes 0-15, 1 for 16-31
    const size_t rowBase = (size_t)blockIdx.x * ROWS;

    if (tid < ROWS) ssq[tid] = 0.f;

    // ---- phase 1: x_tan = clip(artanh(clip(||x||,.99))/||x|| * x, +-50) ----
    for (int rr = 0; rr < ROWS / 8; ++rr) {
        const int r = wave * (ROWS / 8) + rr;        // 8 waves x 8 rows = 64
        const float* xr = x + (rowBase + r) * DIN;
        v4f vals[8];
        float s = 0.f;
        #pragma unroll
        for (int e = 0; e < 8; ++e) {
            v4f v = *(const v4f*)(xr + e * 128 + lane * 4);
            #pragma unroll
            for (int c = 0; c < 4; ++c) {
                float f = v[c];
                f = (f != f) ? 0.f : f;              // nan_to_num
                v[c] = f;
                s += f * f;
            }
            vals[e] = v;
        }
        s = wave_reduce_sum(s);
        const float pn  = fmaxf(sqrtf(s), 1e-15f);
        const float xc  = fminf(pn, 0.99f - 1e-7f);  // clip(.,0.99) then artanh clamp
        const float at  = 0.5f * (log1pf(xc) - log1pf(-xc));
        const float scl = at / pn;
        #pragma unroll
        for (int e = 0; e < 8; ++e) {
            v4h h;
            #pragma unroll
            for (int c = 0; c < 4; ++c) {
                float f = fminf(fmaxf(scl * vals[e][c], -50.f), 50.f);
                h[c] = (_Float16)f;
            }
            *(v4h*)(&xt[r * DIN + e * 128 + lane * 4]) = h;
        }
    }
    __syncthreads();

    // ---- phase 2: WMMA GEMM over 64 col tiles (8 per wave), K = 1024 ----
    float lsum[RT][8];
    #pragma unroll
    for (int rt = 0; rt < RT; ++rt)
        #pragma unroll
        for (int p = 0; p < 8; ++p) lsum[rt][p] = 0.f;

    for (int t8 = 0; t8 < 8; ++t8) {
        const int colBase = (wave * 8 + t8) * 16;
        // B[k][n] = m[n][k]: lane reads 16 consecutive K from m-row (colBase+l15)
        const _Float16* bPtr = mh + (size_t)(colBase + l15) * DIN + h16 * 16;

        v8f acc[RT];
        #pragma unroll
        for (int rt = 0; rt < RT; ++rt) acc[rt] = (v8f){};

        #pragma unroll 2
        for (int ks = 0; ks < 32; ++ks) {
            const int kb = ks * 32;
            v16h b = *(const v16h*)(bPtr + kb);      // shared by 4 row tiles
            #pragma unroll
            for (int rt = 0; rt < RT; ++rt) {
                // A layout: lanes 0-15 -> K {0..15}; lanes 16-31 -> {16..31}
                const _Float16* aRow = &xt[(rt * 16 + l15) * DIN];
                v8h a0 = *(const v8h*)(aRow + kb + h16 * 16);
                v8h a1 = *(const v8h*)(aRow + kb + h16 * 16 + 8);
                v16h a;
                #pragma unroll
                for (int c = 0; c < 8; ++c) { a[c] = a0[c]; a[c + 8] = a1[c]; }
                acc[rt] = __builtin_amdgcn_wmma_f32_16x16x32_f16(
                              false, a, false, b, (short)0, acc[rt], false, false);
            }
        }
        // C layout: VGPR p -> M = p + h16*8, N = l15
        #pragma unroll
        for (int rt = 0; rt < RT; ++rt) {
            #pragma unroll
            for (int p = 0; p < 8; ++p) {
                float v = fminf(fmaxf(acc[rt][p], -1000.f), 1000.f);
                const int rloc = rt * 16 + h16 * 8 + p;
                out[(rowBase + rloc) * (size_t)DIN + colBase + l15] = v;
                lsum[rt][p] += v * v;
            }
        }
    }
    #pragma unroll
    for (int rt = 0; rt < RT; ++rt)
        #pragma unroll
        for (int p = 0; p < 8; ++p)
            atomicAdd(&ssq[rt * 16 + h16 * 8 + p], lsum[rt][p]);  // ds_add_f32
    __syncthreads();

    // ---- phase 3: per-row expmap0 scale (proj is f32 no-op), in LDS ----
    if (tid < ROWS) {
        const float un = fmaxf(sqrtf(ssq[tid]), 1e-15f);
        const float t  = tanhf(un);
        float s = t / un;
        const float mn = 1.0f - 1e-10f;
        if (t > mn) s *= mn / t;
        ssq[tid] = s;                                // reuse as scale array
    }
    __syncthreads();

    // ---- phase 4: rescale the L2-hot mx values this wave just stored ----
    float sc[RT][8];
    #pragma unroll
    for (int rt = 0; rt < RT; ++rt)
        #pragma unroll
        for (int p = 0; p < 8; ++p)
            sc[rt][p] = ssq[rt * 16 + h16 * 8 + p];

    for (int t8 = 0; t8 < 8; ++t8) {
        const int colBase = (wave * 8 + t8) * 16;
        #pragma unroll
        for (int rt = 0; rt < RT; ++rt) {
            #pragma unroll
            for (int p = 0; p < 8; ++p) {
                const int rloc = rt * 16 + h16 * 8 + p;
                const size_t idx =
                    (rowBase + rloc) * (size_t)DIN + colBase + l15;
                out[idx] *= sc[rt][p];   // same-wave store->load: kept in order
            }
        }
    }
}

// ---------------------------------------------------------------------------
extern "C" void kernel_launch(void* const* d_in, const int* in_sizes, int n_in,
                              void* d_out, int out_size, void* d_ws, size_t ws_size,
                              hipStream_t stream) {
    const float* x = (const float*)d_in[0];   // [65536,1024]
    const float* m = (const float*)d_in[1];   // [1024,1024]
    float* out = (float*)d_out;               // [65536,1024]

    _Float16* mh = (_Float16*)d_ws;           // 2 MiB f16 copy of stabilized m

    prep_m     <<<DIN,          256, 0, stream>>>(m, mh);
    mobius_gemm<<<NROWS / ROWS, 256, 0, stream>>>(x, mh, out);
}
